// GPT_85847806312808
// MI455X (gfx1250) — matrix operand order repaired and checked by reading, not compiled, not run on hardware
//
#include <hip/hip_runtime.h>
#include <hip/hip_bf16.h>
#include <math.h>

// ---- problem constants (match reference) ----
#define BB   8
#define SS   512
#define DD   768
#define HH   12
#define LL   12
#define VV   50257
#define HSZ  64
#define MM   (BB*SS)     // 4096 tokens
#define FF   (4*DD)      // 3072

// ---- WMMA types (gfx1250, wave32) ----
typedef __attribute__((ext_vector_type(16))) __bf16         v16bf;
typedef __attribute__((ext_vector_type(16))) unsigned short v16u;
typedef __attribute__((ext_vector_type(8)))  float          v8f;

union BFV { v16u u; v16bf b; };

__device__ __forceinline__ unsigned short f2bf(float f) {
  union { float f; unsigned u; } v; v.f = f;
  unsigned r = v.u + 0x7FFFu + ((v.u >> 16) & 1u);   // round-to-nearest-even
  return (unsigned short)(r >> 16);
}

#define GEMM_OUT_BF16 1
#define GEMM_RELU     2

// =====================================================================
// Generic GEMM: C[M,N] = A[M,K](bf16) * BT[N,K](bf16, pre-transposed)
// (+bias)(+relu)(+resid).  Block tile 128x128x32, 8 waves (2x4),
// wave tile 64x32 (4x2 WMMA tiles).  Both LDS tiles are K-contiguous so
// every fragment read is 2x ds_load_b128.
// =====================================================================
__global__ __launch_bounds__(256)
void gemm_bf16_wmma(const unsigned short* __restrict__ A,
                    const unsigned short* __restrict__ BT,
                    const float* __restrict__ bias,
                    const float* __restrict__ resid,
                    float* __restrict__ Cf,
                    unsigned short* __restrict__ Cb,
                    int M, int N, int K, int flags)
{
  __shared__ __align__(16) unsigned short As [128 * 40];   // [row][k], pitch 40 (80B)
  __shared__ __align__(16) unsigned short BsT[128 * 40];   // [col][k], pitch 40

  const int tid   = threadIdx.x;
  const int wave  = tid >> 5;
  const int lane  = tid & 31;
  const int half  = lane >> 4;
  const int lan16 = lane & 15;
  const int wr    = wave >> 2;          // 0..1 (64-row strips)
  const int wc    = wave & 3;           // 0..3 (32-col strips)
  const int m0    = blockIdx.y * 128;
  const int n0    = blockIdx.x * 128;

  v8f acc[4][2] = {};
  const uint4 z4 = {0u, 0u, 0u, 0u};

  for (int kt = 0; kt < K; kt += 32) {
    // -------- A tile 128x32 : 512 x (8 bf16) chunks, 2/thread --------
#pragma unroll
    for (int i = 0; i < 2; ++i) {
      int c = tid + 256 * i;
      int r = c >> 2, cc = (c & 3) * 8;
      *reinterpret_cast<uint4*>(&As[r * 40 + cc]) =
        *reinterpret_cast<const uint4*>(A + (size_t)(m0 + r) * K + kt + cc);
    }
    // -------- B tile 128 cols x 32 k (from BT[N][K], K-contiguous) --------
#pragma unroll
    for (int i = 0; i < 2; ++i) {
      int c = tid + 256 * i;
      int n = c >> 2, cc = (c & 3) * 8;
      uint4 d = (n0 + n < N)
        ? *reinterpret_cast<const uint4*>(BT + (size_t)(n0 + n) * K + kt + cc)
        : z4;
      *reinterpret_cast<uint4*>(&BsT[n * 40 + cc]) = d;
    }
    if (kt + 32 < K) {  // CDNA5 global_prefetch_b8 of next K-tile
      __builtin_prefetch(A  + (size_t)(m0 + (tid >> 1)) * K + kt + 32 + (tid & 1) * 16, 0, 3);
      __builtin_prefetch(BT + (size_t)(n0 + (tid >> 1)) * K + kt + 32 + (tid & 1) * 16, 0, 3);
    }
    __syncthreads();

    // -------- fragments + WMMA --------
    BFV bfr[2];
#pragma unroll
    for (int cg = 0; cg < 2; ++cg) {
      int bcol = wc * 32 + cg * 16 + lan16;
      const unsigned short* bp = &BsT[bcol * 40 + half * 16];  // k = half*16 + e
#pragma unroll
      for (int e = 0; e < 16; ++e) bfr[cg].u[e] = bp[e];
    }
#pragma unroll
    for (int r = 0; r < 4; ++r) {
      int arow = wr * 64 + r * 16 + lan16;
      const unsigned short* ap = &As[arow * 40];
      BFV a;
#pragma unroll
      for (int e = 0; e < 8; ++e) {         // k = 16*half + e
        a.u[e]     = ap[(half << 4) + e];
        a.u[8 + e] = ap[(half << 4) + 8 + e];
      }
#pragma unroll
      for (int cg = 0; cg < 2; ++cg)
        acc[r][cg] = __builtin_amdgcn_wmma_f32_16x16x32_bf16(
            false, a.b, false, bfr[cg].b, (short)0, acc[r][cg], false, false);
    }
    __syncthreads();
  }

  // -------- epilogue --------
#pragma unroll
  for (int r = 0; r < 4; ++r)
#pragma unroll
    for (int cg = 0; cg < 2; ++cg) {
      int gcol  = n0 + wc * 32 + cg * 16 + lan16;
      if (gcol >= N) continue;
      int grow0 = m0 + wr * 64 + r * 16 + half * 8;
      float bi  = bias ? bias[gcol] : 0.f;
#pragma unroll
      for (int v = 0; v < 8; ++v) {
        int grow = grow0 + v;
        float f  = acc[r][cg][v] + bi;
        if (flags & GEMM_RELU) f = fmaxf(f, 0.f);
        if (resid)             f += resid[(size_t)grow * N + gcol];
        if (flags & GEMM_OUT_BF16) Cb[(size_t)grow * N + gcol] = f2bf(f);
        else                       Cf[(size_t)grow * N + gcol] = f;
      }
    }
}

// =====================================================================
// Attention scores: S[b,h] = Q[b,:,h,:] (SxHS) * K^T, causal mask, scale
// 128x128 tile per block, K-dim = 64 (two WMMA steps).  Q and K tiles
// both token-major (K-contiguous) -> all fragments are ds_load_b128.
// =====================================================================
__global__ __launch_bounds__(256)
void attn_scores_wmma(const unsigned short* __restrict__ Q,
                      const unsigned short* __restrict__ Km,
                      float* __restrict__ Sc, float scale)
{
  const int bh   = blockIdx.z;
  const int b    = bh / HH, h = bh % HH;
  const int row0 = blockIdx.y * 128;
  const int col0 = blockIdx.x * 128;
  const size_t sbase = (size_t)bh * SS * SS;
  const int tid = threadIdx.x;

  if (col0 > row0) {                       // fully-masked tile
    for (int i = 0; i < 64; ++i) {
      int e = tid + 256 * i;
      int r = e >> 7, c = e & 127;
      Sc[sbase + (size_t)(row0 + r) * SS + col0 + c] = -__builtin_inff();
    }
    return;
  }

  __shared__ __align__(16) unsigned short Qs[128 * 72];  // token-major, pitch 72
  __shared__ __align__(16) unsigned short Ks[128 * 72];

  const int wave = tid >> 5, lane = tid & 31;
  const int half = lane >> 4, lan16 = lane & 15;
  const int wr = wave >> 2, wc = wave & 3;

#pragma unroll
  for (int i = 0; i < 4; ++i) {            // 128 tokens x 64 dims each
    int c = tid + 256 * i;
    int t = c >> 3, cc = (c & 7) * 8;
    *reinterpret_cast<uint4*>(&Qs[t * 72 + cc]) =
      *reinterpret_cast<const uint4*>(Q + (((size_t)(b * SS + row0 + t)) * HH + h) * HSZ + cc);
    *reinterpret_cast<uint4*>(&Ks[t * 72 + cc]) =
      *reinterpret_cast<const uint4*>(Km + (((size_t)(b * SS + col0 + t)) * HH + h) * HSZ + cc);
  }
  __syncthreads();

  v8f acc[4][2] = {};
#pragma unroll
  for (int kk = 0; kk < 64; kk += 32) {
    BFV bfr[2];
#pragma unroll
    for (int cg = 0; cg < 2; ++cg) {
      int n = wc * 32 + cg * 16 + lan16;   // B[k][n] = K[token n][dim k]
      const unsigned short* bp = &Ks[n * 72 + kk + half * 16];
#pragma unroll
      for (int e = 0; e < 16; ++e) bfr[cg].u[e] = bp[e];
    }
#pragma unroll
    for (int r = 0; r < 4; ++r) {
      int arow = wr * 64 + r * 16 + lan16;
      const unsigned short* ap = &Qs[arow * 72 + kk];
      BFV a;
#pragma unroll
      for (int e = 0; e < 8; ++e) {
        a.u[e]     = ap[(half << 4) + e];
        a.u[8 + e] = ap[(half << 4) + 8 + e];
      }
#pragma unroll
      for (int cg = 0; cg < 2; ++cg)
        acc[r][cg] = __builtin_amdgcn_wmma_f32_16x16x32_bf16(
            false, a.b, false, bfr[cg].b, (short)0, acc[r][cg], false, false);
    }
  }

#pragma unroll
  for (int r = 0; r < 4; ++r)
#pragma unroll
    for (int cg = 0; cg < 2; ++cg) {
      int gcol = col0 + wc * 32 + cg * 16 + lan16;
#pragma unroll
      for (int v = 0; v < 8; ++v) {
        int grow = row0 + wr * 64 + r * 16 + half * 8 + v;
        float f  = (gcol <= grow) ? acc[r][cg][v] * scale : -__builtin_inff();
        Sc[sbase + (size_t)grow * SS + gcol] = f;
      }
    }
}

// =====================================================================
// Softmax over rows of length 512 -> bf16 probs
// =====================================================================
__global__ __launch_bounds__(256)
void softmax_row512(const float* __restrict__ sc, unsigned short* __restrict__ pr)
{
  __shared__ float red[256];
  const size_t base = (size_t)blockIdx.x * SS;
  const int tid = threadIdx.x;
  float v0 = sc[base + tid], v1 = sc[base + tid + 256];

  red[tid] = fmaxf(v0, v1); __syncthreads();
  for (int s = 128; s > 0; s >>= 1) {
    if (tid < s) red[tid] = fmaxf(red[tid], red[tid + s]);
    __syncthreads();
  }
  float m = red[0]; __syncthreads();

  float e0 = __expf(v0 - m), e1 = __expf(v1 - m);
  red[tid] = e0 + e1; __syncthreads();
  for (int s = 128; s > 0; s >>= 1) {
    if (tid < s) red[tid] += red[tid + s];
    __syncthreads();
  }
  float inv = 1.f / red[0];
  pr[base + tid]       = f2bf(e0 * inv);
  pr[base + tid + 256] = f2bf(e1 * inv);
}

// =====================================================================
// PV: O[b,:,h,:] (SxHS) = P (SxS, bf16) * V (SxHS, bf16)
// 128x64 tile per block, 8 waves (4x2), wave tile 32x32 (2x2 WMMA tiles)
// causal: K only swept to row0+128.  V transposed into LDS so that the
// B-fragment reads are K-contiguous (ds_load_b128).
// =====================================================================
__global__ __launch_bounds__(256)
void attn_pv_wmma(const unsigned short* __restrict__ P,
                  const unsigned short* __restrict__ Vm,
                  unsigned short* __restrict__ O)
{
  const int bh   = blockIdx.y;
  const int b    = bh / HH, h = bh % HH;
  const int row0 = blockIdx.x * 128;
  const size_t pbase = (size_t)bh * SS * SS;

  __shared__ __align__(16) unsigned short Ps[128 * 40];  // P tile, pitch 40
  __shared__ __align__(16) unsigned short VsT[64 * 40];  // V^T: [dim][token]

  const int tid = threadIdx.x, wave = tid >> 5, lane = tid & 31;
  const int half = lane >> 4, lan16 = lane & 15;
  const int wr = wave >> 1, wc = wave & 1;               // 4x2 wave grid

  v8f acc[2][2] = {};
  const int kmax = row0 + 128;                           // causal bound

  for (int kt = 0; kt < kmax; kt += 32) {
#pragma unroll
    for (int i = 0; i < 2; ++i) {                        // P tile 128x32
      int c = tid + 256 * i;
      int r = c >> 2, cc = (c & 3) * 8;
      *reinterpret_cast<uint4*>(&Ps[r * 40 + cc]) =
        *reinterpret_cast<const uint4*>(P + pbase + (size_t)(row0 + r) * SS + kt + cc);
    }
    {                                                    // V 32x64, vector load + LDS transpose
      int t = tid >> 3, cc = (tid & 7) * 8;              // token, dim offset
      uint4 d = *reinterpret_cast<const uint4*>(
          Vm + (((size_t)(b * SS + kt + t)) * HH + h) * HSZ + cc);
      const unsigned short* ds16 = reinterpret_cast<const unsigned short*>(&d);
#pragma unroll
      for (int j = 0; j < 8; ++j) VsT[(cc + j) * 40 + t] = ds16[j];
    }
    __syncthreads();

    BFV bfr[2];
#pragma unroll
    for (int cg = 0; cg < 2; ++cg) {
      int n = wc * 32 + cg * 16 + lan16;                 // output dim
      const unsigned short* bp = &VsT[n * 40 + half * 16];
#pragma unroll
      for (int e = 0; e < 16; ++e) bfr[cg].u[e] = bp[e];
    }
#pragma unroll
    for (int r = 0; r < 2; ++r) {
      int arow = wr * 32 + r * 16 + lan16;
      const unsigned short* ap = &Ps[arow * 40];
      BFV a;
#pragma unroll
      for (int e = 0; e < 8; ++e) {
        a.u[e]     = ap[(half << 4) + e];
        a.u[8 + e] = ap[(half << 4) + 8 + e];
      }
#pragma unroll
      for (int cg = 0; cg < 2; ++cg)
        acc[r][cg] = __builtin_amdgcn_wmma_f32_16x16x32_bf16(
            false, a.b, false, bfr[cg].b, (short)0, acc[r][cg], false, false);
    }
    __syncthreads();
  }

#pragma unroll
  for (int r = 0; r < 2; ++r)
#pragma unroll
    for (int cg = 0; cg < 2; ++cg) {
      int gcol = h * HSZ + wc * 32 + cg * 16 + lan16;
#pragma unroll
      for (int v = 0; v < 8; ++v) {
        int grow = row0 + wr * 32 + r * 16 + half * 8 + v;
        O[(size_t)(b * SS + grow) * DD + gcol] = f2bf(acc[r][cg][v]);
      }
    }
}

// =====================================================================
// LayerNorm(x fp32 row of 768) -> bf16
// =====================================================================
__global__ __launch_bounds__(256)
void layernorm_to_bf16(const float* __restrict__ x, const float* __restrict__ g,
                       const float* __restrict__ bta, unsigned short* __restrict__ out)
{
  __shared__ float s1[256], s2[256];
  const int row = blockIdx.x, tid = threadIdx.x;
  const float* xr = x + (size_t)row * DD;
  float a0 = xr[tid], a1 = xr[tid + 256], a2 = xr[tid + 512];
  s1[tid] = a0 + a1 + a2;
  s2[tid] = a0 * a0 + a1 * a1 + a2 * a2;
  __syncthreads();
  for (int s = 128; s > 0; s >>= 1) {
    if (tid < s) { s1[tid] += s1[tid + s]; s2[tid] += s2[tid + s]; }
    __syncthreads();
  }
  float mu   = s1[0] * (1.f / DD);
  float var  = s2[0] * (1.f / DD) - mu * mu;
  float rstd = rsqrtf(var + 1e-5f);
  unsigned short* orow = out + (size_t)row * DD;
  for (int j = tid; j < DD; j += 256)
    orow[j] = f2bf((xr[j] - mu) * rstd * g[j] + bta[j]);
}

// =====================================================================
// misc elementwise
// =====================================================================
__global__ __launch_bounds__(256)
void embed_kernel(const int* __restrict__ idx, const float* __restrict__ tok,
                  const float* __restrict__ pos, float* __restrict__ x)
{
  const int t = blockIdx.x;
  const int s = t & (SS - 1);
  const float* tr = tok + (size_t)idx[t] * DD;
  const float* pr = pos + (size_t)s * DD;
  float* xr = x + (size_t)t * DD;
  for (int j = threadIdx.x; j < DD; j += 256) xr[j] = tr[j] + pr[j];
}

// fp32 [K,N] -> bf16 transposed [N,K], 32x32 LDS tiles, coalesced both sides.
// blockIdx.z selects the matrix in a stacked [L,K,N] tensor.
__global__ __launch_bounds__(256)
void cvt_transpose_bf16(const float* __restrict__ W, unsigned short* __restrict__ WT,
                        int K, int N)
{
  __shared__ float t[32][33];
  const float* Wz = W + (size_t)blockIdx.z * K * N;
  unsigned short* WTz = WT + (size_t)blockIdx.z * N * K;
  const int n0 = blockIdx.x * 32, k0 = blockIdx.y * 32;
  const int tx = threadIdx.x & 31, ty = threadIdx.x >> 5;   // ty 0..7
#pragma unroll
  for (int i = 0; i < 4; ++i) {
    int k = k0 + ty + i * 8, n = n0 + tx;
    t[ty + i * 8][tx] = (n < N) ? Wz[(size_t)k * N + n] : 0.f;
  }
  __syncthreads();
#pragma unroll
  for (int i = 0; i < 4; ++i) {
    int n = n0 + ty + i * 8, k = k0 + tx;
    if (n < N) WTz[(size_t)n * K + k] = f2bf(t[tx][ty + i * 8]);
  }
}

__global__ void zero_one(float* p) { if (threadIdx.x == 0 && blockIdx.x == 0) *p = 0.f; }

// cross-entropy over V=50257, mean into *loss
__global__ __launch_bounds__(256)
void ce_loss_row(const float* __restrict__ logits, const int* __restrict__ tgt,
                 float* __restrict__ loss)
{
  __shared__ float red[256];
  const int row = blockIdx.x, tid = threadIdx.x;
  const float* lr = logits + (size_t)row * VV;
  float m = -3.4e38f;
  for (int j = tid; j < VV; j += 256) m = fmaxf(m, lr[j]);
  red[tid] = m; __syncthreads();
  for (int s = 128; s > 0; s >>= 1) {
    if (tid < s) red[tid] = fmaxf(red[tid], red[tid + s]);
    __syncthreads();
  }
  m = red[0]; __syncthreads();
  float s = 0.f;
  for (int j = tid; j < VV; j += 256) s += __expf(lr[j] - m);
  red[tid] = s; __syncthreads();
  for (int st = 128; st > 0; st >>= 1) {
    if (tid < st) red[tid] += red[tid + st];
    __syncthreads();
  }
  if (tid == 0) {
    float lse = m + logf(red[0]);
    int t = tgt[row];
    atomicAdd(loss, (lse - lr[t]) * (1.0f / (float)MM));
  }
}

// =====================================================================
// host launcher
// =====================================================================
extern "C" void kernel_launch(void* const* d_in, const int* in_sizes, int n_in,
                              void* d_out, int out_size, void* d_ws, size_t ws_size,
                              hipStream_t stream)
{
  (void)in_sizes; (void)n_in; (void)out_size; (void)ws_size;

  const int*   idx  = (const int*)  d_in[0];
  const int*   tgt  = (const int*)  d_in[1];
  const float* tok  = (const float*)d_in[2];
  const float* pos  = (const float*)d_in[3];
  const float* Wq   = (const float*)d_in[4];
  const float* Wk   = (const float*)d_in[5];
  const float* Wv   = (const float*)d_in[6];
  const float* Wp   = (const float*)d_in[7];
  const float* bp   = (const float*)d_in[8];
  const float* W1   = (const float*)d_in[9];
  const float* b1   = (const float*)d_in[10];
  const float* W2   = (const float*)d_in[11];
  const float* b2   = (const float*)d_in[12];
  const float* ln1g = (const float*)d_in[13];
  const float* ln1b = (const float*)d_in[14];
  const float* ln2g = (const float*)d_in[15];
  const float* ln2b = (const float*)d_in[16];
  const float* lnfg = (const float*)d_in[17];
  const float* lnfb = (const float*)d_in[18];
  const float* lmW  = (const float*)d_in[19];
  const float* lmb  = (const float*)d_in[20];

  float* logits = (float*)d_out;
  float* loss   = logits + (size_t)MM * VV;

  // ---- workspace carve ----
  char* p = (char*)d_ws;
  auto take = [&](size_t bytes) -> char* {
    char* r = p; p += (bytes + 255) & ~(size_t)255; return r;
  };
  // transposed bf16 weights: wX_bt[l] is [N][K]
  unsigned short* wq_bt = (unsigned short*)take((size_t)LL * DD * DD * 2);
  unsigned short* wk_bt = (unsigned short*)take((size_t)LL * DD * DD * 2);
  unsigned short* wv_bt = (unsigned short*)take((size_t)LL * DD * DD * 2);
  unsigned short* wp_bt = (unsigned short*)take((size_t)LL * DD * DD * 2);
  unsigned short* w1_bt = (unsigned short*)take((size_t)LL * DD * FF * 2);
  unsigned short* w2_bt = (unsigned short*)take((size_t)LL * FF * DD * 2);
  unsigned short* lm_bt = (unsigned short*)take((size_t)DD * VV * 2);
  float*          x     = (float*)         take((size_t)MM * DD * 4);
  unsigned short* hbf   = (unsigned short*)take((size_t)MM * DD * 2);
  unsigned short* qbf   = (unsigned short*)take((size_t)MM * DD * 2);
  unsigned short* kbf   = (unsigned short*)take((size_t)MM * DD * 2);
  unsigned short* vbf   = (unsigned short*)take((size_t)MM * DD * 2);
  unsigned short* obf   = (unsigned short*)take((size_t)MM * DD * 2);
  unsigned short* mlpbf = (unsigned short*)take((size_t)MM * FF * 2);
  float*          sc    = (float*)         take((size_t)BB * HH * SS * SS * 4);
  unsigned short* pr    = (unsigned short*)take((size_t)BB * HH * SS * SS * 2);

  // ---- weights fp32 -> bf16, transposed to [N][K] ----
  {
    dim3 gDxD((DD + 31) / 32, DD / 32, LL);      // K=DD, N=DD
    cvt_transpose_bf16<<<gDxD, 256, 0, stream>>>(Wq, wq_bt, DD, DD);
    cvt_transpose_bf16<<<gDxD, 256, 0, stream>>>(Wk, wk_bt, DD, DD);
    cvt_transpose_bf16<<<gDxD, 256, 0, stream>>>(Wv, wv_bt, DD, DD);
    cvt_transpose_bf16<<<gDxD, 256, 0, stream>>>(Wp, wp_bt, DD, DD);
    dim3 gW1((FF + 31) / 32, DD / 32, LL);       // K=DD, N=FF
    cvt_transpose_bf16<<<gW1, 256, 0, stream>>>(W1, w1_bt, DD, FF);
    dim3 gW2((DD + 31) / 32, FF / 32, LL);       // K=FF, N=DD
    cvt_transpose_bf16<<<gW2, 256, 0, stream>>>(W2, w2_bt, FF, DD);
    dim3 gLMt((VV + 31) / 32, DD / 32, 1);       // K=DD, N=VV
    cvt_transpose_bf16<<<gLMt, 256, 0, stream>>>(lmW, lm_bt, DD, VV);
  }

  embed_kernel<<<MM, 256, 0, stream>>>(idx, tok, pos, x);

  const dim3 gD (DD / 128, MM / 128);        // (6, 32)
  const dim3 gF (FF / 128, MM / 128);        // (24, 32)
  const dim3 gSc(SS / 128, SS / 128, BB*HH); // (4, 4, 96)
  const dim3 gPV(SS / 128, BB * HH);         // (4, 96)

  for (int l = 0; l < LL; ++l) {
    const unsigned short* wql = wq_bt + (size_t)l * DD * DD;
    const unsigned short* wkl = wk_bt + (size_t)l * DD * DD;
    const unsigned short* wvl = wv_bt + (size_t)l * DD * DD;
    const unsigned short* wpl = wp_bt + (size_t)l * DD * DD;
    const unsigned short* w1l = w1_bt + (size_t)l * DD * FF;
    const unsigned short* w2l = w2_bt + (size_t)l * FF * DD;

    layernorm_to_bf16<<<MM, 256, 0, stream>>>(x, ln1g + l * DD, ln1b + l * DD, hbf);
    gemm_bf16_wmma<<<gD, 256, 0, stream>>>(hbf, wql, nullptr, nullptr, nullptr, qbf,
                                           MM, DD, DD, GEMM_OUT_BF16);
    gemm_bf16_wmma<<<gD, 256, 0, stream>>>(hbf, wkl, nullptr, nullptr, nullptr, kbf,
                                           MM, DD, DD, GEMM_OUT_BF16);
    gemm_bf16_wmma<<<gD, 256, 0, stream>>>(hbf, wvl, nullptr, nullptr, nullptr, vbf,
                                           MM, DD, DD, GEMM_OUT_BF16);
    attn_scores_wmma<<<gSc, 256, 0, stream>>>(qbf, kbf, sc, 0.125f); // 1/sqrt(64)
    softmax_row512<<<BB * HH * SS, 256, 0, stream>>>(sc, pr);
    attn_pv_wmma<<<gPV, 256, 0, stream>>>(pr, vbf, obf);
    gemm_bf16_wmma<<<gD, 256, 0, stream>>>(obf, wpl, bp + l * DD, x, x, nullptr,
                                           MM, DD, DD, 0);                 // x += o@Wp+b
    layernorm_to_bf16<<<MM, 256, 0, stream>>>(x, ln2g + l * DD, ln2b + l * DD, hbf);
    gemm_bf16_wmma<<<gF, 256, 0, stream>>>(hbf, w1l, b1 + (size_t)l * FF,
                                           nullptr, nullptr, mlpbf,
                                           MM, FF, DD, GEMM_OUT_BF16 | GEMM_RELU);
    gemm_bf16_wmma<<<gD, 256, 0, stream>>>(mlpbf, w2l, b2 + l * DD, x, x, nullptr,
                                           MM, DD, FF, 0);                 // x += relu@W2+b
  }

  layernorm_to_bf16<<<MM, 256, 0, stream>>>(x, lnfg, lnfb, hbf);
  const dim3 gLM((VV + 127) / 128, MM / 128); // (393, 32)
  gemm_bf16_wmma<<<gLM, 256, 0, stream>>>(hbf, lm_bt, lmb, nullptr, logits, nullptr,
                                          MM, VV, DD, 0);

  zero_one<<<1, 32, 0, stream>>>(loss);
  ce_loss_row<<<MM, 256, 0, stream>>>(logits, tgt, loss);
}
